// MultilayerGRU_56066503082285
// MI455X (gfx1250) — matrix operand order repaired and checked by reading, not compiled, not run on hardware
//
#include <hip/hip_runtime.h>
#include <cmath>

// ---------------------------------------------------------------------------
// Problem constants (match reference)
// ---------------------------------------------------------------------------
#define GB 64      // batch
#define GS 512     // seq len
#define GI 128     // input dim
#define GH 1024    // hidden dim
#define GL 3       // layers
#define GO 128     // output dim

typedef __attribute__((ext_vector_type(16))) __bf16 v16bf;
typedef __attribute__((ext_vector_type(8)))  float  v8f;
typedef __attribute__((ext_vector_type(4)))  unsigned int uint4v;

union FragBF { v16bf v; uint4v u[2]; };

static __device__ __forceinline__ float bf2f(unsigned short h) {
    union { unsigned int u; float f; } c; c.u = ((unsigned int)h) << 16; return c.f;
}
static __device__ __forceinline__ unsigned short f2bf(float f) {
    union { float f; unsigned int u; } c; c.f = f;
    unsigned int u = c.u;
    u += 0x7FFFu + ((u >> 16) & 1u);   // round-to-nearest-even
    return (unsigned short)(u >> 16);
}
static __device__ __forceinline__ float sigmoidf_(float x) {
    return 1.0f / (1.0f + __expf(-x));
}

// ---------------------------------------------------------------------------
// One K=32 WMMA step: A-fragment (16xK activations) x 4 B-fragments (weights)
// A layout per CDNA5 ISA 16-bit A 16x32; B: lane n holds the same K order as A of
// weight row (n0+n) (weights stored row-major [N,K] == B column-major).
// ---------------------------------------------------------------------------
static __device__ __forceinline__ void mma_step(v8f acc[4],
                                                const unsigned short* Arow,
                                                const unsigned short* W, int ldw,
                                                int k0, int kh, int l15) {
    FragBF a;
    a.u[0] = *(const uint4v*)(Arow + k0 + 8 * kh);
    a.u[1] = *(const uint4v*)(Arow + k0 + 16 + 8 * kh);
#pragma unroll
    for (int i = 0; i < 4; ++i) {
        const unsigned short* wp = W + (size_t)(i * 16 + l15) * ldw + k0 + 8 * kh;
        FragBF b;
        b.u[0] = *(const uint4v*)(wp);
        b.u[1] = *(const uint4v*)(wp + 16);
        acc[i] = __builtin_amdgcn_wmma_f32_16x16x32_bf16(
            false, a.v, false, b.v, (short)0, acc[i], false, false);
    }
}

// ---------------------------------------------------------------------------
// Pipelined slot kernel. Slot s runs stage (layer l, phase p=s&1) at
// t = (s - 2l - p)/2.  Phase 0 (zr): P = inp@Wx[z,r].T + h@Wh[z,r].T + b ->
// z = sig, rh = sig(r)*h.  Phase 1 (g): G = inp@Wxg.T + rh@Whg.T + b ->
// h' = z*h + (1-z)*tanh(G).
// Grid: 48 blocks x 256 threads = 384 waves = 3 layers x 128 waves.
// ---------------------------------------------------------------------------
__global__ __launch_bounds__(256) void gru_slot_kernel(
    int s,
    const unsigned short* __restrict__ xbf,   // [B,S,I] bf16
    const unsigned short* __restrict__ wx0,   // [3H,I]
    const unsigned short* __restrict__ wh0,   // [3H,H]
    const unsigned short* __restrict__ wxr,   // [2,3H,H]
    const unsigned short* __restrict__ whr,   // [2,3H,H]
    const float* __restrict__ bh0,            // [3H]
    const float* __restrict__ bhr,            // [2,3H]
    unsigned short* __restrict__ h01,         // 4 slots of [B,H]: L0{0,1}, L1{0,1}
    unsigned short* __restrict__ hist,        // [(S+1),B,H] layer-2 history
    float* __restrict__ zbuf,                 // [3][B,H]
    unsigned short* __restrict__ rhbuf)       // [3][B,H]
{
    const int p    = s & 1;
    const int wave = (blockIdx.x * blockDim.x + threadIdx.x) >> 5;
    const int lane = threadIdx.x & 31;
    const int layer = wave >> 7;            // 128 waves per layer
    const int rem   = wave & 127;
    const int mt    = rem & 3;              // 4 M-tiles (B=64)
    const int ng    = rem >> 2;             // N-groups of 64 cols
    if (layer >= GL) return;
    if (p == 1 && ng >= 16) return;         // g-phase: only 1024 cols
    const int t = (s - 2 * layer - p) >> 1;
    if (t < 0 || t >= GS) return;

    const int l15 = lane & 15, kh = lane >> 4;
    const int m0 = mt * 16;
    const int mrow = m0 + l15;

    // Layer-specific operand selection
    const unsigned short *Wx, *Wh, *inp, *hprev;
    const float* bias;
    int Kin; long lda_in;
    if (layer == 0) {
        Wx = wx0; Wh = wh0; bias = bh0; Kin = GI;
        inp = xbf + (size_t)t * GI;          lda_in = (long)GS * GI;
        hprev = h01 + (size_t)(t & 1) * (GB * GH);
    } else {
        Wx = wxr + (size_t)(layer - 1) * 3 * GH * GH;
        Wh = whr + (size_t)(layer - 1) * 3 * GH * GH;
        bias = bhr + (size_t)(layer - 1) * 3 * GH;
        Kin = GH; lda_in = GH;
        if (layer == 1) {
            inp   = h01 + (size_t)((t + 1) & 1) * (GB * GH);       // h0(t)
            hprev = h01 + (size_t)(2 + (t & 1)) * (GB * GH);       // h1(t-1)
        } else {
            inp   = h01 + (size_t)(2 + ((t + 1) & 1)) * (GB * GH); // h1(t)
            hprev = hist + (size_t)t * (GB * GH);                  // h2(t-1)
        }
    }

    const int n0 = ng * 64;
    const int wrow0 = (p == 0) ? n0 : (2 * GH + n0);  // g-gate rows at [2H,3H)

    v8f acc[4];
#pragma unroll
    for (int i = 0; i < 4; ++i) acc[i] = (v8f){0.f,0.f,0.f,0.f,0.f,0.f,0.f,0.f};

    const unsigned short* Arow1 = inp + (size_t)mrow * lda_in;
    const unsigned short* A2    = (p == 0) ? hprev : (rhbuf + (size_t)layer * GB * GH);
    const unsigned short* Arow2 = A2 + (size_t)mrow * GH;
    const unsigned short* Wxp   = Wx + (size_t)wrow0 * Kin;
    const unsigned short* Whp   = Wh + (size_t)wrow0 * GH;

    for (int k0 = 0; k0 < Kin; k0 += 32) mma_step(acc, Arow1, Wxp, Kin, k0, kh, l15);
    for (int k0 = 0; k0 < GH;  k0 += 32) mma_step(acc, Arow2, Whp, GH,  k0, kh, l15);

    if (p == 0) {
        // zr epilogue: cols [0,2H): z=sig -> zbuf ; r=sig -> rh=r*h(bf16)
        float* zb = zbuf + (size_t)layer * GB * GH;
        unsigned short* rb = rhbuf + (size_t)layer * GB * GH;
#pragma unroll
        for (int i = 0; i < 4; ++i) {
#pragma unroll
            for (int j = 0; j < 8; ++j) {
                int col = n0 + i * 16 + l15;
                int row = m0 + j + 8 * kh;
                float v = acc[i][j] + bias[col];
                float sg = sigmoidf_(v);
                if (col < GH) {
                    zb[(size_t)row * GH + col] = sg;
                } else {
                    int c = col - GH;
                    float hv = bf2f(hprev[(size_t)row * GH + c]);
                    rb[(size_t)row * GH + c] = f2bf(sg * hv);
                }
            }
        }
    } else {
        // g epilogue: h' = z*h + (1-z)*tanh(G)
        unsigned short* hdst;
        if (layer == 0)      hdst = h01 + (size_t)((t + 1) & 1) * (GB * GH);
        else if (layer == 1) hdst = h01 + (size_t)(2 + ((t + 1) & 1)) * (GB * GH);
        else                 hdst = hist + (size_t)(t + 1) * (GB * GH);
        const float* zb = zbuf + (size_t)layer * GB * GH;
#pragma unroll
        for (int i = 0; i < 4; ++i) {
#pragma unroll
            for (int j = 0; j < 8; ++j) {
                int col = n0 + i * 16 + l15;
                int row = m0 + j + 8 * kh;
                float v = acc[i][j] + bias[2 * GH + col];
                float g = tanhf(v);
                float z = zb[(size_t)row * GH + col];
                float hv = bf2f(hprev[(size_t)row * GH + col]);
                hdst[(size_t)row * GH + col] = f2bf(z * hv + (1.0f - z) * g);
            }
        }
    }
}

// ---------------------------------------------------------------------------
// Final output projection: y[b,t,:] = h2(t) @ Wout.T + bout
// A = hist slots 1..S flattened as [S*B, H]; 4096 waves x (16x64) tiles.
// ---------------------------------------------------------------------------
__global__ __launch_bounds__(256) void gru_outproj_kernel(
    const unsigned short* __restrict__ hist,
    const unsigned short* __restrict__ woutbf,   // [O,H]
    const float* __restrict__ bout,
    float* __restrict__ out)                     // [B,S,O]
{
    const int wave = (blockIdx.x * blockDim.x + threadIdx.x) >> 5;
    const int lane = threadIdx.x & 31;
    const int l15 = lane & 15, kh = lane >> 4;
    const int ng = wave & 1;          // 2 groups of 64 cols (O=128)
    const int mt = wave >> 1;         // (S*B)/16 = 2048 M-tiles
    if (mt >= (GS * GB) / 16) return;
    const int m0 = mt * 16;
    const int mrow = m0 + l15;

    v8f acc[4];
#pragma unroll
    for (int i = 0; i < 4; ++i) acc[i] = (v8f){0.f,0.f,0.f,0.f,0.f,0.f,0.f,0.f};

    const unsigned short* Arow = hist + (size_t)(GB * GH) /*skip t=-1 slot*/ + (size_t)mrow * GH;
    const unsigned short* Wp = woutbf + (size_t)(ng * 64) * GH;
    for (int k0 = 0; k0 < GH; k0 += 32) mma_step(acc, Arow, Wp, GH, k0, kh, l15);

#pragma unroll
    for (int i = 0; i < 4; ++i) {
#pragma unroll
        for (int j = 0; j < 8; ++j) {
            int col = ng * 64 + i * 16 + l15;
            int r = m0 + j + 8 * kh;        // r = t*B + b
            int tt = r >> 6, b = r & 63;
            out[(size_t)b * GS * GO + (size_t)tt * GO + col] = acc[i][j] + bout[col];
        }
    }
}

// Final hidden state [B, L, H] (fp32) from bf16 states
__global__ void gru_hcopy_kernel(const unsigned short* __restrict__ h01,
                                 const unsigned short* __restrict__ hist,
                                 float* __restrict__ out2) {
    int idx = blockIdx.x * blockDim.x + threadIdx.x;
    if (idx >= GB * GL * GH) return;
    int i = idx & (GH - 1);
    int l = (idx >> 10) % GL;
    int b = idx / (GL * GH);
    const int fs = GS & 1;   // final write slot parity
    const unsigned short* src;
    if (l == 0)      src = h01 + (size_t)fs * (GB * GH) + (size_t)b * GH + i;
    else if (l == 1) src = h01 + (size_t)(2 + fs) * (GB * GH) + (size_t)b * GH + i;
    else             src = hist + (size_t)GS * (GB * GH) + (size_t)b * GH + i;
    out2[idx] = bf2f(*src);
}

// ---------------------------------------------------------------------------
// Init helpers
// ---------------------------------------------------------------------------
__global__ void cvt_f32_bf16_kernel(const float* __restrict__ src,
                                    unsigned short* __restrict__ dst, long n) {
    long i = (long)blockIdx.x * blockDim.x + threadIdx.x;
    long stride = (long)gridDim.x * blockDim.x;
    for (; i < n; i += stride) dst[i] = f2bf(src[i]);
}
__global__ void zero_us_kernel(unsigned short* __restrict__ p, long n) {
    long i = (long)blockIdx.x * blockDim.x + threadIdx.x;
    long stride = (long)gridDim.x * blockDim.x;
    for (; i < n; i += stride) p[i] = 0;
}

// ---------------------------------------------------------------------------
// Host launcher
// ---------------------------------------------------------------------------
extern "C" void kernel_launch(void* const* d_in, const int* in_sizes, int n_in,
                              void* d_out, int out_size, void* d_ws, size_t ws_size,
                              hipStream_t stream) {
    const float* x    = (const float*)d_in[0];
    const float* Wx0  = (const float*)d_in[1];
    const float* Wh0  = (const float*)d_in[2];
    const float* bh0  = (const float*)d_in[3];
    const float* Wxr  = (const float*)d_in[4];
    const float* Whr  = (const float*)d_in[5];
    const float* bhr  = (const float*)d_in[6];
    const float* Wout = (const float*)d_in[7];
    const float* bout = (const float*)d_in[8];
    float* out = (float*)d_out;

    const long NX   = (long)GB * GS * GI;
    const long NWX0 = 3L * GH * GI;
    const long NWH0 = 3L * GH * GH;
    const long NWXR = 2L * 3 * GH * GH;
    const long NWHR = NWXR;
    const long NWOUT = (long)GO * GH;

    // workspace layout (256B aligned regions)
    char* base = (char*)d_ws;
    size_t off = 0;
    auto alloc = [&](size_t bytes) { size_t o = off; off = (off + bytes + 255) & ~(size_t)255; return o; };
    unsigned short* xbf   = (unsigned short*)(base + alloc(NX * 2));
    unsigned short* wx0b  = (unsigned short*)(base + alloc(NWX0 * 2));
    unsigned short* wh0b  = (unsigned short*)(base + alloc(NWH0 * 2));
    unsigned short* wxrb  = (unsigned short*)(base + alloc(NWXR * 2));
    unsigned short* whrb  = (unsigned short*)(base + alloc(NWHR * 2));
    unsigned short* woutb = (unsigned short*)(base + alloc(NWOUT * 2));
    unsigned short* h01   = (unsigned short*)(base + alloc((size_t)4 * GB * GH * 2));
    unsigned short* hist  = (unsigned short*)(base + alloc((size_t)(GS + 1) * GB * GH * 2));
    float*          zbuf  = (float*)        (base + alloc((size_t)GL * GB * GH * 4));
    unsigned short* rhbuf = (unsigned short*)(base + alloc((size_t)GL * GB * GH * 2));
    (void)ws_size; (void)n_in; (void)in_sizes; (void)out_size;

    // 1) convert inputs/weights to bf16
    cvt_f32_bf16_kernel<<<2048, 256, 0, stream>>>(x, xbf, NX);
    cvt_f32_bf16_kernel<<<1024, 256, 0, stream>>>(Wx0, wx0b, NWX0);
    cvt_f32_bf16_kernel<<<2048, 256, 0, stream>>>(Wh0, wh0b, NWH0);
    cvt_f32_bf16_kernel<<<4096, 256, 0, stream>>>(Wxr, wxrb, NWXR);
    cvt_f32_bf16_kernel<<<4096, 256, 0, stream>>>(Whr, whrb, NWHR);
    cvt_f32_bf16_kernel<<<512, 256, 0, stream>>>(Wout, woutb, NWOUT);

    // 2) zero initial hidden states (h0/h1 ping-pong slot 0 + h2 history slot 0)
    zero_us_kernel<<<256, 256, 0, stream>>>(h01, (long)4 * GB * GH);
    zero_us_kernel<<<256, 256, 0, stream>>>(hist, (long)GB * GH);

    // 3) pipelined recurrence: slot s runs (layer l, phase p=s&1) at t=(s-2l-p)/2
    const int nslots = 2 * GS + 4;
    for (int s = 0; s < nslots; ++s) {
        gru_slot_kernel<<<48, 256, 0, stream>>>(
            s, xbf, wx0b, wh0b, wxrb, whrb, bh0, bhr, h01, hist, zbuf, rhbuf);
    }

    // 4) output projection over full h2 history: [S*B,H] x [H,O]
    gru_outproj_kernel<<<512, 256, 0, stream>>>(hist, woutb, bout, out);

    // 5) final hidden state [B,L,H] fp32 appended after layer_output
    float* out2 = out + (size_t)GB * GS * GO;
    gru_hcopy_kernel<<<(GB * GL * GH + 255) / 256, 256, 0, stream>>>(h01, hist, out2);
}
